// MultiHeadAttention_32770600468859
// MI455X (gfx1250) — matrix operand order repaired and checked by reading, not compiled, not run on hardware
//
#include <hip/hip_runtime.h>
#include <hip/hip_bf16.h>

// Problem constants (from reference): B=2, S=1024, D=1024, H=16, DK=64
#define B_  2
#define S_  1024
#define D_  1024
#define H_  16
#define DK_ 64
#define M_  (B_*S_)   // 2048 rows of activations

typedef __attribute__((ext_vector_type(16))) _Float16 v16h;
typedef __attribute__((ext_vector_type(8)))  _Float16 v8h;
typedef __attribute__((ext_vector_type(8)))  float    v8f;

#ifndef __has_builtin
#define __has_builtin(x) 0
#endif

// gfx1250 async global->LDS path (ASYNCcnt), guarded so both toolchains build.
#if defined(__AMDGCN__) && __has_builtin(__builtin_amdgcn_global_load_async_to_lds_b128)
#define HAVE_ASYNC_LDS 1
#else
#define HAVE_ASYNC_LDS 0
#endif

#if HAVE_ASYNC_LDS
// Builtin signature (from hipcc diagnostic): param 0 is
// 'int __vector(4) __device__ *'  == v4i in addrspace(1); LDS side addrspace(3).
typedef int v4i __attribute__((vector_size(16)));
typedef __attribute__((address_space(1))) v4i* glb_v4i_p;
typedef __attribute__((address_space(3))) v4i* lds_v4i_p;
#endif

__device__ __forceinline__ void wait_async0() {
#if HAVE_ASYNC_LDS
#if __has_builtin(__builtin_amdgcn_s_wait_asynccnt)
  __builtin_amdgcn_s_wait_asynccnt(0);
#else
  asm volatile("s_wait_asynccnt 0" ::: "memory");
#endif
#endif
}

// ---------------------------------------------------------------------------
// Fragment builders for 16-bit WMMA operands; A and B share a contiguous K order (K = 16*(L/16) + s).
//
// A (16x32, f16): lane L holds row M=L%16.
//   lanes 0-15 : halves 0..15 = K 0..15
//   lanes 16-31: halves 0..15 = K 16..31
__device__ __forceinline__ v16h make_a_frag(const _Float16* rowbase, int kk, int laneHi) {
  const v8h x0 = *(const v8h*)(rowbase + kk + laneHi * 16);
  const v8h x1 = *(const v8h*)(rowbase + kk + laneHi * 16 + 8);
  v16h r;
#pragma unroll
  for (int i = 0; i < 8; ++i) { r[i] = x0[i]; r[8 + i] = x1[i]; }
  return r;
}

// B (32x16, f16): lane L holds column N=L%16; 16 contiguous K values
// starting at K = 16*(L/16): one contiguous 32B load from a K-contiguous row.
__device__ __forceinline__ v16h load_b_frag(const _Float16* p) {
  return *(const v16h*)p;
}

#define WMMA_F16(a, b, c) \
  __builtin_amdgcn_wmma_f32_16x16x32_f16(false, (a), false, (b), (short)0, (c), false, false)

// ---------------------------------------------------------------------------
// Kernel 1: Y = X @ W^T + bias, output f16 in head-split layout.
//   z=0: Q -> qh[b,h,s,dk]   z=1: K -> kh[b,h,s,dk]   z=2: V -> vhT[b,h,dk,s]
// Block: 256 thr (8 waves). Tile: BM=128, BN=64, BK=64. Each wave owns a
// 32x32 output tile -> 8 WMMAs per K-step with 2x A/B fragment reuse.
__global__ __launch_bounds__(256)
void proj_qkv_kernel(const float* __restrict__ q, const float* __restrict__ k,
                     const float* __restrict__ v,
                     const float* __restrict__ Wq, const float* __restrict__ Wk,
                     const float* __restrict__ Wv,
                     const float* __restrict__ bq, const float* __restrict__ bk,
                     const float* __restrict__ bv,
                     _Float16* __restrict__ qh, _Float16* __restrict__ kh,
                     _Float16* __restrict__ vhT)
{
  const int z = blockIdx.z;
  const float* X    = (z == 0) ? q  : (z == 1) ? k  : v;
  const float* W    = (z == 0) ? Wq : (z == 1) ? Wk : Wv;
  const float* bias = (z == 0) ? bq : (z == 1) ? bk : bv;
  _Float16*    O    = (z == 0) ? qh : (z == 1) ? kh : vhT;
  const int vmode = (z == 2);

  const int n0 = blockIdx.x * 64;
  const int m0 = blockIdx.y * 128;
  const int tid = threadIdx.x;
  const int lane = tid & 31, wv = tid >> 5;
  const int laneHi = lane >> 4, lane16 = lane & 15;
  const int m_w = wv & 3;          // 4 stripes of 32 rows
  const int n_w = wv >> 2;         // 2 stripes of 32 cols

  __shared__ alignas(32) _Float16 lds_a[128][64];  // 16 KB
  __shared__ alignas(32) _Float16 lds_b[64][64];   //  8 KB

  v8f acc[2][2] = {};

  const int srow = tid >> 3;        // 0..31
  const int scol = (tid & 7) * 8;   // 0..56

  for (int k0 = 0; k0 < D_; k0 += 64) {
    // stage A (128x64) and B (64x64) as f16 (fp32 -> f16 convert on the fly)
#pragma unroll
    for (int rr = 0; rr < 4; ++rr) {
      const float* pa = X + (m0 + srow + rr * 32) * D_ + k0 + scol;
#pragma unroll
      for (int i = 0; i < 8; ++i) lds_a[srow + rr * 32][scol + i] = (_Float16)pa[i];
    }
#pragma unroll
    for (int rr = 0; rr < 2; ++rr) {
      const float* pb = W + (n0 + srow + rr * 32) * D_ + k0 + scol;
#pragma unroll
      for (int i = 0; i < 8; ++i) lds_b[srow + rr * 32][scol + i] = (_Float16)pb[i];
    }
    __syncthreads();

#pragma unroll
    for (int kc = 0; kc < 64; kc += 32) {
      const v16h a0 = make_a_frag(&lds_a[m_w * 32 + lane16][0],      kc, laneHi);
      const v16h a1 = make_a_frag(&lds_a[m_w * 32 + 16 + lane16][0], kc, laneHi);
      const v16h b0 = load_b_frag(&lds_b[n_w * 32 + lane16][kc + laneHi * 16]);
      const v16h b1 = load_b_frag(&lds_b[n_w * 32 + 16 + lane16][kc + laneHi * 16]);
      acc[0][0] = WMMA_F16(a0, b0, acc[0][0]);
      acc[0][1] = WMMA_F16(a0, b1, acc[0][1]);
      acc[1][0] = WMMA_F16(a1, b0, acc[1][0]);
      acc[1][1] = WMMA_F16(a1, b1, acc[1][1]);
    }
    __syncthreads();
  }

  // D-fragment: VGPR r holds row M = r + 8*(L/16), column N = L%16.
#pragma unroll
  for (int mt = 0; mt < 2; ++mt) {
#pragma unroll
    for (int nt = 0; nt < 2; ++nt) {
      const int mg  = m0 + m_w * 32 + mt * 16 + laneHi * 8;
      const int col = n0 + n_w * 32 + nt * 16 + lane16;
      const float bcol = bias[col];
      const int h = col >> 6, dk = col & 63;
#pragma unroll
      for (int r = 0; r < 8; ++r) {
        const int m  = mg + r;
        const int bb = m >> 10;          // m / S_
        const int s  = m & (S_ - 1);
        const float val = acc[mt][nt][r] + bcol;
        if (!vmode) O[((bb * H_ + h) * S_ + s) * DK_ + dk] = (_Float16)val;
        else        O[((bb * H_ + h) * DK_ + dk) * S_ + s] = (_Float16)val;
      }
    }
  }
}

// ---------------------------------------------------------------------------
// Kernel 2: attention for one (16-query tile, head, batch) per block.
__global__ __launch_bounds__(256)
void attn_kernel(const _Float16* __restrict__ qh, const _Float16* __restrict__ kh,
                 const _Float16* __restrict__ vhT, const int* __restrict__ mask,
                 _Float16* __restrict__ ctx)
{
  const int q0 = blockIdx.x * 16;
  const int h  = blockIdx.y;
  const int b  = blockIdx.z;
  const int bh = b * H_ + h;
  const int tid = threadIdx.x;
  const int lane = tid & 31, wv = tid >> 5;
  const int laneHi = lane >> 4, lane16 = lane & 15;

  __shared__ alignas(32) _Float16 sc[16][S_];   // 32 KB: scores -> probs (f16)
  __shared__ float part[2][16][DK_];            //  8 KB: split-K partials
  __shared__ float red[16][16];                 //  1 KB: softmax reductions

  // ---- Phase 1: scores = (qh @ kh^T) / sqrt(DK), masked ----
  const _Float16* qrow = qh + (bh * S_ + q0 + lane16) * DK_;
  const v16h qf0 = make_a_frag(qrow, 0, laneHi);   // dk 0..31
  const v16h qf1 = make_a_frag(qrow, 32, laneHi);  // dk 32..63

#pragma unroll
  for (int t = 0; t < 8; ++t) {
    const int n0 = wv * 128 + t * 16;            // key-tile base
    const _Float16* krow = kh + (bh * S_ + n0 + lane16) * DK_;
    const v16h kf0 = load_b_frag(krow + laneHi * 16);
    const v16h kf1 = load_b_frag(krow + 32 + laneHi * 16);
    v8f acc = {};
    acc = WMMA_F16(qf0, kf0, acc);
    acc = WMMA_F16(qf1, kf1, acc);
    const int col = n0 + lane16;
    const int mk  = mask[b * S_ + col];
#pragma unroll
    for (int r = 0; r < 8; ++r) {
      float vsc = acc[r] * 0.125f;               // 1/sqrt(64)
      if (mk == 0) vsc = -60000.0f;              // NEG, clamped to f16 range
      sc[r + laneHi * 8][col] = (_Float16)vsc;
    }
  }
  __syncthreads();

  // ---- Phase 2: row softmax (fp32 math, f16 storage) ----
  const int srow = tid >> 4;    // 0..15
  const int ssub = tid & 15;    // 0..15, 64 columns each
  {
    float pm = -3.0e38f;
    for (int c = ssub * 64; c < ssub * 64 + 64; ++c)
      pm = fmaxf(pm, (float)sc[srow][c]);
    red[srow][ssub] = pm;
  }
  __syncthreads();
  float rowmax = -3.0e38f;
#pragma unroll
  for (int i = 0; i < 16; ++i) rowmax = fmaxf(rowmax, red[srow][i]);
  __syncthreads();
  {
    float ps = 0.0f;
    for (int c = ssub * 64; c < ssub * 64 + 64; ++c) {
      const float e = __expf((float)sc[srow][c] - rowmax);
      sc[srow][c] = (_Float16)e;
      ps += e;
    }
    red[srow][ssub] = ps;
  }
  __syncthreads();
  float rs = 0.0f;
#pragma unroll
  for (int i = 0; i < 16; ++i) rs += red[srow][i];
  const float inv = 1.0f / rs;
  for (int c = ssub * 64; c < ssub * 64 + 64; ++c)
    sc[srow][c] = (_Float16)((float)sc[srow][c] * inv);
  __syncthreads();

  // ---- Phase 3: out = attn @ vh (split-K=2 over the 1024 keys) ----
  const int dkt   = wv & 3;           // which 16-wide dk tile
  const int kbase = (wv >> 2) * 512;  // K half
  const _Float16* vrow = vhT + (bh * DK_ + dkt * 16 + lane16) * S_;
  v8f acc = {};
  for (int kk = kbase; kk < kbase + 512; kk += 32) {
    const v16h a   = make_a_frag(&sc[lane16][0], kk, laneHi);
    const v16h bfr = load_b_frag(vrow + kk + laneHi * 16);
    acc = WMMA_F16(a, bfr, acc);
  }
#pragma unroll
  for (int r = 0; r < 8; ++r)
    part[wv >> 2][r + laneHi * 8][dkt * 16 + lane16] = acc[r];
  __syncthreads();

  // reduce the two K-halves and write f16 context, row-major [B,S,D]
  {
    const int row = tid >> 4;
    const int c4  = (tid & 15) * 4;
#pragma unroll
    for (int i = 0; i < 4; ++i) {
      const float s2 = part[0][row][c4 + i] + part[1][row][c4 + i];
      ctx[(b * S_ + q0 + row) * D_ + h * DK_ + c4 + i] = (_Float16)s2;
    }
  }
}

// ---------------------------------------------------------------------------
// Kernel 3: out = ctx @ Wo^T + bo (f16 A via async global->LDS, fp32 W
// converted, fp32 output). Same BM=128/BN=64/BK=64 tiling as kernel 1.
__global__ __launch_bounds__(256)
void out_proj_kernel(const _Float16* __restrict__ ctx, const float* __restrict__ Wo,
                     const float* __restrict__ bo, float* __restrict__ out)
{
  const int n0 = blockIdx.x * 64;
  const int m0 = blockIdx.y * 128;
  const int tid = threadIdx.x;
  const int lane = tid & 31, wv = tid >> 5;
  const int laneHi = lane >> 4, lane16 = lane & 15;
  const int m_w = wv & 3;
  const int n_w = wv >> 2;

  __shared__ alignas(32) _Float16 lds_a[128][64];  // 16 KB (f16 ctx tile)
  __shared__ alignas(32) _Float16 lds_b[64][64];   //  8 KB (f16 Wo tile)

  v8f acc[2][2] = {};

  const int srow = tid >> 3;        // 0..31
  const int scol = (tid & 7) * 8;   // 0..56

  for (int k0 = 0; k0 < D_; k0 += 64) {
    // A tile: 128x64 halfs = 1024 x 16B chunks; 4 chunks per thread.
#pragma unroll
    for (int i = 0; i < 4; ++i) {
      const int chunk = tid + i * 256;
      const int row = chunk >> 3;
      const int col = (chunk & 7) * 8;
      const _Float16* src = ctx + (m0 + row) * D_ + k0 + col;
#if HAVE_ASYNC_LDS
      __builtin_amdgcn_global_load_async_to_lds_b128(
          (glb_v4i_p)(void*)src, (lds_v4i_p)(void*)&lds_a[row][col], 0, 0);
#else
      *(v8h*)&lds_a[row][col] = *(const v8h*)src;
#endif
    }
    // B tile: Wo fp32 -> f16 convert while staging.
#pragma unroll
    for (int rr = 0; rr < 2; ++rr) {
      const float* pb = Wo + (n0 + srow + rr * 32) * D_ + k0 + scol;
#pragma unroll
      for (int i = 0; i < 8; ++i) lds_b[srow + rr * 32][scol + i] = (_Float16)pb[i];
    }
    wait_async0();
    __syncthreads();

#pragma unroll
    for (int kc = 0; kc < 64; kc += 32) {
      const v16h a0 = make_a_frag(&lds_a[m_w * 32 + lane16][0],      kc, laneHi);
      const v16h a1 = make_a_frag(&lds_a[m_w * 32 + 16 + lane16][0], kc, laneHi);
      const v16h b0 = load_b_frag(&lds_b[n_w * 32 + lane16][kc + laneHi * 16]);
      const v16h b1 = load_b_frag(&lds_b[n_w * 32 + 16 + lane16][kc + laneHi * 16]);
      acc[0][0] = WMMA_F16(a0, b0, acc[0][0]);
      acc[0][1] = WMMA_F16(a0, b1, acc[0][1]);
      acc[1][0] = WMMA_F16(a1, b0, acc[1][0]);
      acc[1][1] = WMMA_F16(a1, b1, acc[1][1]);
    }
    __syncthreads();
  }

#pragma unroll
  for (int mt = 0; mt < 2; ++mt) {
#pragma unroll
    for (int nt = 0; nt < 2; ++nt) {
      const int mg  = m0 + m_w * 32 + mt * 16 + laneHi * 8;
      const int col = n0 + n_w * 32 + nt * 16 + lane16;
      const float bcol = bo[col];
#pragma unroll
      for (int r = 0; r < 8; ++r)
        out[(mg + r) * D_ + col] = acc[mt][nt][r] + bcol;
    }
  }
}

// ---------------------------------------------------------------------------
extern "C" void kernel_launch(void* const* d_in, const int* in_sizes, int n_in,
                              void* d_out, int out_size, void* d_ws, size_t ws_size,
                              hipStream_t stream)
{
  (void)in_sizes; (void)n_in; (void)out_size; (void)ws_size;
  const float* q    = (const float*)d_in[0];
  const float* k    = (const float*)d_in[1];
  const float* v    = (const float*)d_in[2];
  const int*   mask = (const int*)  d_in[3];
  const float* Wq   = (const float*)d_in[4];
  const float* bq   = (const float*)d_in[5];
  const float* Wk   = (const float*)d_in[6];
  const float* bk   = (const float*)d_in[7];
  const float* Wv   = (const float*)d_in[8];
  const float* bv   = (const float*)d_in[9];
  const float* Wo   = (const float*)d_in[10];
  const float* bo   = (const float*)d_in[11];
  float* out = (float*)d_out;

  // Workspace: 4 f16 tensors of B*H*S*DK = 2M elements (4 MB each, 16 MB total)
  char* ws = (char*)d_ws;
  const size_t bytes = (size_t)B_ * H_ * S_ * DK_ * sizeof(_Float16);
  _Float16* qh  = (_Float16*)(ws + 0 * bytes);
  _Float16* kh  = (_Float16*)(ws + 1 * bytes);
  _Float16* vhT = (_Float16*)(ws + 2 * bytes);
  _Float16* ctx = (_Float16*)(ws + 3 * bytes);

  const dim3 blk(256);
  proj_qkv_kernel<<<dim3(D_ / 64, M_ / 128, 3), blk, 0, stream>>>(
      q, k, v, Wq, Wk, Wv, bq, bk, bv, qh, kh, vhT);
  attn_kernel<<<dim3(S_ / 16, H_, B_), blk, 0, stream>>>(qh, kh, vhT, mask, ctx);
  out_proj_kernel<<<dim3(D_ / 64, M_ / 128, 1), blk, 0, stream>>>(ctx, Wo, bo, out);
}
